// MultiHeadTimeMixing_18004502905108
// MI455X (gfx1250) — matrix operand order repaired and checked by reading, not compiled, not run on hardware
//
#include <hip/hip_runtime.h>
#include <hip/hip_bf16.h>

// ---------------- problem constants (match reference) ----------------
#define T_ 2048
#define E_ 1024
#define H_ 8
#define HE_ (H_ * E_)   // 8192
#define WTILES 4        // N-direction 16x16 C-tiles per wave
#define MTILES 2        // M-direction 16-row blocks per wave

typedef __attribute__((ext_vector_type(16))) __bf16 v16bf;
typedef __attribute__((ext_vector_type(16))) unsigned short v16u;
typedef __attribute__((ext_vector_type(8))) float v8f;

__device__ __forceinline__ unsigned short f2bf(float f) {
    unsigned u = __builtin_bit_cast(unsigned, f);
    // round-to-nearest-even truncation to bf16
    unsigned r = (u + 0x7FFFu + ((u >> 16) & 1u)) >> 16;
    return (unsigned short)r;
}

// ---------------- 1) token shift + time-mix (f32 -> bf16) ----------------
__global__ __launch_bounds__(256) void prep_kernel(
    const float* __restrict__ x, const float* __restrict__ sx,
    const float* __restrict__ tmk, const float* __restrict__ tmv,
    const float* __restrict__ tmr,
    unsigned short* __restrict__ kx, unsigned short* __restrict__ vx,
    unsigned short* __restrict__ rx) {
    int idx = blockIdx.x * blockDim.x + threadIdx.x;   // [0, T_*E_)
    int t = idx / E_;
    int e = idx - t * E_;
    float xe   = x[idx];
    float prev = (t == 0) ? sx[e] : x[idx - E_];
    float mk = tmk[e], mv = tmv[e], mr = tmr[e];
    kx[idx] = f2bf(xe * mk + prev * (1.0f - mk));
    vx[idx] = f2bf(xe * mv + prev * (1.0f - mv));
    rx[idx] = f2bf(xe * mr + prev * (1.0f - mr));
}

// ---------------- 2) f32 -> bf16 + pack into WMMA B-fragment layout -------
// For (kTile, nTile) 32x16 block of B[K,N], lane l (0..31) of the consuming
// wave needs values[i] = B[kTile*32 + (l>>4)*16 + i][nTile*16 + (l&15)],
// i = 0..15.  Stored contiguously at out[((kTile*nTiles + nTile)*32 + l)*16 + i]
// so the GEMM fetches each fragment with two fully-coalesced b128 loads.
__global__ __launch_bounds__(256) void pack_B_kernel(
    const float* __restrict__ W, unsigned short* __restrict__ out,
    int K, int N) {
    int tid  = blockIdx.x * blockDim.x + threadIdx.x;
    int lane = tid & 31;
    int frag = tid >> 5;
    int nTiles = N >> 4;
    int kTiles = K >> 5;
    if (frag >= kTiles * nTiles) return;
    int kt = frag / nTiles;
    int nt = frag - kt * nTiles;
    int half = lane >> 4;
    int l16  = lane & 15;
    int col  = nt * 16 + l16;
    union { unsigned short u[16]; uint4 q[2]; } tmp;
#pragma unroll
    for (int i = 0; i < 16; ++i) {
        tmp.u[i] = f2bf(W[(size_t)(kt * 32 + half * 16 + i) * N + col]);
    }
    uint4* dst = (uint4*)(out + ((size_t)frag * 32 + lane) * 16);
    dst[0] = tmp.q[0];
    dst[1] = tmp.q[1];
}

// ---------------- 3) WMMA bf16 GEMM:  C[M,N] = act(A @ B) (+residual) -----
// A row-major bf16 [M,K]; Bp fragment-packed (see pack_B_kernel).
// 8 waves/block; each wave computes a (MTILES*16) x (WTILES*16) block of C,
// reusing each B fragment for MTILES A fragments (register blocking).
// act: 0 = none, 1 = sigmoid. residual may be nullptr.
__global__ __launch_bounds__(256) void gemm_bf16_kernel(
    const unsigned short* __restrict__ A,
    const unsigned short* __restrict__ Bp,
    float* __restrict__ C,
    const float* __restrict__ residual,
    int M, int N, int K, int act) {
    const int wave = threadIdx.x >> 5;
    const int lane = threadIdx.x & 31;
    const int half = lane >> 4;
    const int l16  = lane & 15;
    const int nTiles = N >> 4;
    const int tile0  = (blockIdx.x * 8 + wave) * WTILES;  // first N-tile
    const int tileM  = blockIdx.y * (MTILES * 16);        // first row
    if (tile0 >= nTiles || tileM >= M) return;

    v8f acc[MTILES][WTILES];
#pragma unroll
    for (int m = 0; m < MTILES; ++m)
#pragma unroll
        for (int j = 0; j < WTILES; ++j) acc[m][j] = (v8f){};

    const int kTiles = K >> 5;
    for (int kt = 0; kt < kTiles; ++kt) {
        const int kk = kt << 5;
        // A fragments (one per 16-row block): row = l16 (+m*16);
        //   values[0..15] at K = kk + half*16 + j   (two b128 loads each)
        v16bf a[MTILES];
#pragma unroll
        for (int m = 0; m < MTILES; ++m) {
            const uint4* ap = (const uint4*)(
                A + (size_t)(tileM + m * 16 + l16) * K + kk + half * 16);
            union { uint4 q[2]; v16u v; } aU;
            aU.q[0] = ap[0];
            aU.q[1] = ap[1];   // +8 elements = +16 bytes
            a[m] = __builtin_bit_cast(v16bf, aU.v);
        }
        // B fragments: packed contiguously; one lane reads 32B per tile.
        const uint4* bp = (const uint4*)(
            Bp + (((size_t)kt * nTiles + tile0) * 32 + lane) * 16);
#pragma unroll
        for (int j = 0; j < WTILES; ++j) {
            union { uint4 q[2]; v16u v; } bU;
            const uint4* bj = bp + j * 64;   // 512 elems per tile-frag
            bU.q[0] = bj[0];
            bU.q[1] = bj[1];
            v16bf b = __builtin_bit_cast(v16bf, bU.v);
#pragma unroll
            for (int m = 0; m < MTILES; ++m) {
                acc[m][j] = __builtin_amdgcn_wmma_f32_16x16x32_bf16(
                    false, a[m], false, b, (short)0, acc[m][j], false, false);
            }
        }
    }

    // C layout: VGPR i -> row M = i (+8 for lanes 16-31), col N = l16
#pragma unroll
    for (int m = 0; m < MTILES; ++m) {
#pragma unroll
        for (int j = 0; j < WTILES; ++j) {
            int col = (tile0 + j) * 16 + l16;
#pragma unroll
            for (int i = 0; i < 8; ++i) {
                int row = tileM + m * 16 + i + (half ? 8 : 0);
                float v = acc[m][j][i];
                if (act == 1) v = 1.0f / (1.0f + __expf(-v));
                size_t o = (size_t)row * N + col;
                if (residual) v += residual[o];
                C[o] = v;
            }
        }
    }
}

// ---------------- 4) WKV sequential scan (parallel over H*E channels) -----
__global__ __launch_bounds__(256) void wkv_scan_kernel(
    const float* __restrict__ k, const float* __restrict__ v,
    const float* __restrict__ aa0, const float* __restrict__ bb0,
    const float* __restrict__ pp0,
    const float* __restrict__ time_first, const float* __restrict__ time_decay,
    float* __restrict__ wkv, float* __restrict__ out_states /* 3*HE_ */) {
    int c = blockIdx.x * blockDim.x + threadIdx.x;   // [0, HE_)
    float aa = aa0[c], bb = bb0[c], pp = pp0[c];
    const float tf = time_first[c];
    const float wd = __expf(time_decay[c]);
    // software pipeline: preload next token's k/v while computing current
    float kk = k[c];
    float vv = v[c];
    for (int t = 0; t < T_; ++t) {
        float kn = 0.0f, vn = 0.0f;
        if (t + 1 < T_) {
            size_t nidx = (size_t)(t + 1) * HE_ + c;
            kn = k[nidx];
            vn = v[nidx];
        }
        float ww = tf + kk;
        float q  = fmaxf(pp, ww);
        float e1 = __expf(pp - q);
        float e2 = __expf(ww - q);
        wkv[(size_t)t * HE_ + c] = (e1 * aa + e2 * vv) / (e1 * bb + e2);
        float ww2 = pp - wd;
        float qp  = fmaxf(ww2, kk);
        e1 = __expf(ww2 - qp);
        e2 = __expf(kk - qp);
        aa = e1 * aa + e2 * vv;
        bb = e1 * bb + e2;
        pp = qp;
        kk = kn;
        vv = vn;
    }
    out_states[c]           = aa;
    out_states[HE_ + c]     = bb;
    out_states[2 * HE_ + c] = pp;
}

// ---------------- 5) head-collapse mix -> bf16 ----------------
__global__ __launch_bounds__(256) void mix_kernel(
    const float* __restrict__ r, const float* __restrict__ wkv,
    const float* __restrict__ head_collapse,
    unsigned short* __restrict__ mixed) {
    int idx = blockIdx.x * blockDim.x + threadIdx.x;  // [0, T_*E_)
    int t = idx / E_;
    int e = idx - t * E_;
    float hcv[H_];
    float m = -3.4e38f;
#pragma unroll
    for (int h = 0; h < H_; ++h) { hcv[h] = head_collapse[h]; m = fmaxf(m, hcv[h]); }
    float s = 0.0f;
#pragma unroll
    for (int h = 0; h < H_; ++h) { hcv[h] = __expf(hcv[h] - m); s += hcv[h]; }
    float inv = 1.0f / s;
    float acc = 0.0f;
#pragma unroll
    for (int h = 0; h < H_; ++h) {
        float hc = hcv[h] * inv;
        size_t o = (size_t)t * HE_ + (size_t)h * E_ + e;
        acc += hc * hc * r[o] * wkv[o];
    }
    mixed[idx] = f2bf(acc);
}

// ---------------- 6) copy last token embedding into output tuple ----------
__global__ __launch_bounds__(256) void last_row_kernel(
    const float* __restrict__ x, float* __restrict__ dst) {
    int e = blockIdx.x * blockDim.x + threadIdx.x;
    if (e < E_) dst[e] = x[(size_t)(T_ - 1) * E_ + e];
}

// =================== host launch ===================
extern "C" void kernel_launch(void* const* d_in, const int* in_sizes, int n_in,
                              void* d_out, int out_size, void* d_ws, size_t ws_size,
                              hipStream_t stream) {
    const float* x           = (const float*)d_in[0];
    const float* sx          = (const float*)d_in[1];
    const float* aa          = (const float*)d_in[2];
    const float* bb          = (const float*)d_in[3];
    const float* pp          = (const float*)d_in[4];
    const float* time_first  = (const float*)d_in[5];
    const float* time_decay  = (const float*)d_in[6];
    const float* time_mix_k  = (const float*)d_in[7];
    const float* time_mix_v  = (const float*)d_in[8];
    const float* time_mix_r  = (const float*)d_in[9];
    const float* W_k         = (const float*)d_in[10];
    const float* W_v         = (const float*)d_in[11];
    const float* W_r         = (const float*)d_in[12];
    const float* W_out       = (const float*)d_in[13];
    const float* head_coll   = (const float*)d_in[14];

    float* out      = (float*)d_out;                     // [T,E]
    float* x_last   = out + (size_t)T_ * E_;             // [E]
    float* states   = x_last + E_;                       // aa,bb,pp each [HE_]

    // ---- carve workspace ----
    char* wsp = (char*)d_ws;
    auto carve = [&](size_t bytes) -> char* {
        char* p = wsp;
        wsp += (bytes + 255) & ~(size_t)255;
        return p;
    };
    unsigned short* kx   = (unsigned short*)carve((size_t)T_ * E_ * 2);
    unsigned short* vx   = (unsigned short*)carve((size_t)T_ * E_ * 2);
    unsigned short* rx   = (unsigned short*)carve((size_t)T_ * E_ * 2);
    unsigned short* Wkb  = (unsigned short*)carve((size_t)E_ * HE_ * 2);
    unsigned short* Wvb  = (unsigned short*)carve((size_t)E_ * HE_ * 2);
    unsigned short* Wrb  = (unsigned short*)carve((size_t)E_ * HE_ * 2);
    unsigned short* Wob  = (unsigned short*)carve((size_t)E_ * E_ * 2);
    float* kbuf          = (float*)carve((size_t)T_ * HE_ * 4);
    float* vbuf          = (float*)carve((size_t)T_ * HE_ * 4);
    float* rbuf          = (float*)carve((size_t)T_ * HE_ * 4);
    float* wkvbuf        = (float*)carve((size_t)T_ * HE_ * 4);
    unsigned short* mix  = (unsigned short*)carve((size_t)T_ * E_ * 2);
    (void)ws_size; (void)in_sizes; (void)n_in; (void)out_size;

    const int TE = T_ * E_;

    // 1) token shift + time mix
    prep_kernel<<<TE / 256, 256, 0, stream>>>(x, sx, time_mix_k, time_mix_v,
                                              time_mix_r, kx, vx, rx);

    // 2) weight conversion + B-fragment packing (f32 -> bf16)
    {
        int frags = (E_ / 32) * (HE_ / 16);     // 32 * 512
        int thr   = frags * 32;                 // 524288
        pack_B_kernel<<<(thr + 255) / 256, 256, 0, stream>>>(W_k, Wkb, E_, HE_);
        pack_B_kernel<<<(thr + 255) / 256, 256, 0, stream>>>(W_v, Wvb, E_, HE_);
        pack_B_kernel<<<(thr + 255) / 256, 256, 0, stream>>>(W_r, Wrb, E_, HE_);
        int fragsO = (E_ / 32) * (E_ / 16);     // 32 * 64
        int thrO   = fragsO * 32;               // 65536
        pack_B_kernel<<<(thrO + 255) / 256, 256, 0, stream>>>(W_out, Wob, E_, E_);
    }

    // 3) big GEMMs: [T,E] x [E,HE] -> [T,HE]
    //    each block: 8 waves * WTILES(4) * 16 = 512 cols, MTILES*16 = 32 rows
    dim3 g1(HE_ / (8 * WTILES * 16), T_ / (MTILES * 16));   // (16, 64)
    dim3 blk(256);
    gemm_bf16_kernel<<<g1, blk, 0, stream>>>(kx, Wkb, kbuf, nullptr,
                                             T_, HE_, E_, /*act=*/0);
    gemm_bf16_kernel<<<g1, blk, 0, stream>>>(vx, Wvb, vbuf, nullptr,
                                             T_, HE_, E_, /*act=*/0);
    gemm_bf16_kernel<<<g1, blk, 0, stream>>>(rx, Wrb, rbuf, nullptr,
                                             T_, HE_, E_, /*act=*/1);

    // 4) WKV scan (parallel over 8192 channels)
    wkv_scan_kernel<<<HE_ / 256, 256, 0, stream>>>(
        kbuf, vbuf, aa, bb, pp, time_first, time_decay, wkvbuf, states);

    // 5) head-collapse softmax mix
    mix_kernel<<<TE / 256, 256, 0, stream>>>(rbuf, wkvbuf, head_coll, mix);

    // 6) out = x + mixed @ W_out
    dim3 g2(E_ / (8 * WTILES * 16), T_ / (MTILES * 16));    // (2, 64)
    gemm_bf16_kernel<<<g2, blk, 0, stream>>>(mix, Wob, out, x,
                                             T_, E_, E_, /*act=*/0);

    // last token embedding
    last_row_kernel<<<(E_ + 255) / 256, 256, 0, stream>>>(x, x_last);
}
